// MinkowskiEncoder_32658931318802
// MI455X (gfx1250) — matrix operand order repaired and checked by reading, not compiled, not run on hardware
//
#include <hip/hip_runtime.h>
#include <hip/hip_bf16.h>

typedef __attribute__((ext_vector_type(16))) _Float16 v16h;
typedef __attribute__((ext_vector_type(8)))  _Float16 h8;
typedef __attribute__((ext_vector_type(8)))  float    v8f;
typedef __attribute__((ext_vector_type(4)))  float    f4;

#define NEGBIG (-1e30f)
#define BN_EPS (1e-5f)

// ---------------------------------------------------------------------------
// Masked 3x3 stride-2 maxpool (pad 1). If maskIn == nullptr, activity is
// derived from x > 0 (level-0 depth sparsification; C==1 there).
// ---------------------------------------------------------------------------
__global__ void maxpool_kernel(const float* __restrict__ x,
                               const float* __restrict__ maskIn,
                               float* __restrict__ xo,
                               float* __restrict__ masko,
                               int B, int Hi, int Wi, int Ho, int Wo, int C) {
  long long idx = (long long)blockIdx.x * blockDim.x + threadIdx.x;
  long long total = (long long)B * Ho * Wo * C;
  if (idx >= total) return;
  int c = (int)(idx % C);
  long long p = idx / C;
  int wo = (int)(p % Wo);
  long long t = p / Wo;
  int ho = (int)(t % Ho);
  int b  = (int)(t / Ho);

  float best = NEGBIG;
  float m = 0.0f;
  for (int i = 0; i < 3; ++i) {
    int hi = 2 * ho + i - 1;
    if (hi < 0 || hi >= Hi) continue;
    for (int j = 0; j < 3; ++j) {
      int wi = 2 * wo + j - 1;
      if (wi < 0 || wi >= Wi) continue;
      long long q = ((long long)(b * Hi + hi) * Wi + wi);
      float v  = x[q * C + c];
      float mm = maskIn ? maskIn[q] : (v > 0.0f ? 1.0f : 0.0f);
      if (mm > 0.0f) {
        m = 1.0f;
        best = fmaxf(best, v);
      }
    }
  }
  xo[idx] = (m > 0.0f) ? best : 0.0f;
  if (c == 0) masko[p] = m;
}

// ---------------------------------------------------------------------------
// n = sum(mask). Single block, LDS tree reduction -> deterministic.
// ---------------------------------------------------------------------------
__global__ void mask_count_kernel(const float* __restrict__ mask, int npix,
                                  float* __restrict__ nbuf) {
  __shared__ float s[256];
  float acc = 0.0f;
  for (int i = threadIdx.x; i < npix; i += 256) acc += mask[i];
  s[threadIdx.x] = acc;
  __syncthreads();
  for (int o = 128; o > 0; o >>= 1) {
    if ((int)threadIdx.x < o) s[threadIdx.x] += s[threadIdx.x + o];
    __syncthreads();
  }
  if (threadIdx.x == 0) nbuf[0] = s[0];
}

// ---------------------------------------------------------------------------
// Per-channel sum / sum-of-squares. Inputs are already masked (zero at
// inactive sites): sum(x*mask)==sum(x), sum(x^2*mask)==sum(x^2).
// ---------------------------------------------------------------------------
__global__ void bn_reduce_kernel(const float* __restrict__ x, int npix, int C,
                                 float* __restrict__ sumb,
                                 float* __restrict__ sumsqb) {
  int c = blockIdx.x;
  __shared__ float s1[256];
  __shared__ float s2[256];
  float a = 0.0f, q = 0.0f;
  for (long long i = threadIdx.x; i < npix; i += 256) {
    float v = x[i * C + c];
    a += v;
    q += v * v;
  }
  s1[threadIdx.x] = a;
  s2[threadIdx.x] = q;
  __syncthreads();
  for (int o = 128; o > 0; o >>= 1) {
    if ((int)threadIdx.x < o) {
      s1[threadIdx.x] += s1[threadIdx.x + o];
      s2[threadIdx.x] += s2[threadIdx.x + o];
    }
    __syncthreads();
  }
  if (threadIdx.x == 0) {
    sumb[c] = s1[0];
    sumsqb[c] = s2[0];
  }
}

// ---------------------------------------------------------------------------
// y = relu((x - mean) * rsqrt(var+eps) * gamma + beta) * mask
// ---------------------------------------------------------------------------
__global__ void bn_norm_relu_kernel(const float* __restrict__ x,
                                    const float* __restrict__ mask,
                                    const float* __restrict__ g,
                                    const float* __restrict__ bta,
                                    const float* __restrict__ sumb,
                                    const float* __restrict__ sumsqb,
                                    const float* __restrict__ nbuf,
                                    float* __restrict__ out,
                                    long long npix, int C) {
  long long idx = (long long)blockIdx.x * blockDim.x + threadIdx.x;
  long long total = npix * C;
  if (idx >= total) return;
  int c = (int)(idx % C);
  long long p = idx / C;
  float n = nbuf[0];
  float mean = sumb[c] / n;
  float var = sumsqb[c] / n - mean * mean;
  float inv = rsqrtf(var + BN_EPS);
  float y = (x[idx] - mean) * inv * g[c] + bta[c];
  y = y > 0.0f ? y : 0.0f;
  out[idx] = y * mask[p];
}

// ---------------------------------------------------------------------------
// Repack HWIO f32 weights into fragment-ready f16:
//   fragment f = (tap*NCH + chunk)*NTN + tn, 32 lanes x 16 halfs each.
//   element e of lane -> K = (lane>>4)*16 + e, ci = chunk*32 + K,
//   co = tn*16 + (lane&15). A wave's B fragment = two b128 loads.
// ---------------------------------------------------------------------------
__global__ void weight_swizzle_kernel(const float* __restrict__ w,
                                      _Float16* __restrict__ wsw,
                                      int in_c, int out_c, int taps) {
  int nch = (in_c + 31) / 32;
  int ntn = out_c >> 4;
  long long total = (long long)taps * nch * ntn * 512;
  long long i = (long long)blockIdx.x * blockDim.x + threadIdx.x;
  if (i >= total) return;
  int e = (int)(i & 15);
  long long t = i >> 4;
  int lane = (int)(t & 31);
  t >>= 5;
  int tn = (int)(t % ntn);
  t /= ntn;
  int ch = (int)(t % nch);
  int tap = (int)(t / nch);
  int ci = ch * 32 + (lane >> 4) * 16 + e;
  int co = tn * 16 + (lane & 15);
  float v = (ci < in_c) ? w[((long long)tap * in_c + ci) * out_c + co] : 0.0f;
  wsw[i] = (_Float16)v;
}

// ---------------------------------------------------------------------------
// Masked im2col conv as WMMA GEMM. Each wave owns one 16-pixel m-tile and
// TWO 16-wide n-tiles (out_c is always a multiple of 32), so every A gather
// feeds two v_wmma. The 8 waves of a block share the n-tiles, so the current
// tap's B fragments are staged once per block into LDS (<=16KB) and read
// back as ds_load_b128. IC/KK are compile-time -> chunk loops unroll.
//
// A layout (16-bit A 16x32): row M = lane&15; lane<16 holds K 0..15,
// lane>=16 holds K 16..31 (contiguous order, as in the B swizzle) -> one run of 16 consecutive channels
// per lane -> 4 x b128 global loads. D: VGPR r -> M = r+8*(lane>=16),
// N = lane&15. No early exit: tail waves run on zero A fragments so all
// threads reach the staging barriers.
// ---------------------------------------------------------------------------
template <int IC, int KKT>
__global__ __launch_bounds__(256) void conv_wmma_kernel(
    const float* __restrict__ x, const float* __restrict__ mask,
    const _Float16* __restrict__ wsw, float* __restrict__ out,
    int npix, int H, int W, int out_c, int accumulate) {
  constexpr int NCH = (IC + 31) / 32;
  constexpr int PAD = KKT >> 1;
  __shared__ _Float16 bsh[2 * NCH * 512];

  const int lane = threadIdx.x & 31;
  const int waveid = threadIdx.x >> 5;
  const int tile_m = blockIdx.x * 8 + waveid;
  const int tn0 = blockIdx.y * 2;  // two adjacent n-tiles per wave
  const int ntn = out_c >> 4;

  const int row = lane & 15;
  const int hi16 = lane >> 4;

  const int p = tile_m * 16 + row;
  const bool prow = p < npix;  // also false for whole tail waves
  int b, h, ww;
  {
    int pp = prow ? p : 0;
    b = pp / (H * W);
    int r = pp % (H * W);
    h = r / W;
    ww = r % W;
  }

  // Prefetch the kk input rows of this pixel's neighborhood.
  for (int ky = 0; ky < KKT; ++ky) {
    int hh = h + ky - PAD;
    if (prow && hh >= 0 && hh < H) {
      int w0 = ww - PAD;
      if (w0 < 0) w0 = 0;
      __builtin_prefetch(x + ((long long)(b * H + hh) * W + w0) * IC, 0, 1);
    }
  }

  v8f c0 = {}, c1 = {};
  for (int ky = 0; ky < KKT; ++ky) {
    int hh = h + ky - PAD;
    bool hok = prow && (hh >= 0) && (hh < H);
    for (int kx = 0; kx < KKT; ++kx) {
      const int tap = ky * KKT + kx;

      // ---- stage this tap's B fragments (both n-tiles) into LDS ----
      for (int i = threadIdx.x; i < 2 * NCH * 64; i += 256) {
        int off = i & 63;
        int lf = i >> 6;          // 0..2*NCH-1
        int ch = lf >> 1;
        int tsel = lf & 1;
        ((h8*)bsh)[(tsel * NCH + ch) * 64 + off] =
            *(const h8*)(wsw +
                         ((long long)(tap * NCH + ch) * ntn + tn0 + tsel) *
                             512 +
                         off * 8);
      }
      __syncthreads();

      int wx = ww + kx - PAD;
      bool ok = hok && (wx >= 0) && (wx < W);
      const float* ap =
          x + (ok ? ((long long)(b * H + hh) * W + wx) * IC : 0);
#pragma unroll
      for (int ch = 0; ch < NCH; ++ch) {
        // ---- A fragment (shared by both n-tiles) ----
        v16h a = {};
        if constexpr (IC == 1) {
          float v = 0.0f;
          if (ok && hi16 == 0) v = ap[0];  // only K==0 is a real channel
          a[0] = (_Float16)v;
        } else {
          if (ok) {
            const float* r0 = ap + ch * 32 + hi16 * 16;      // K 0..7 / 16..23
            const float* r1 = ap + ch * 32 + hi16 * 16 + 8;  // K 8..15/24..31
            f4 x0 = *(const f4*)(r0);
            f4 x1 = *(const f4*)(r0 + 4);
            f4 x2 = *(const f4*)(r1);
            f4 x3 = *(const f4*)(r1 + 4);
#pragma unroll
            for (int e = 0; e < 4; ++e) {
              a[e]      = (_Float16)x0[e];
              a[4 + e]  = (_Float16)x1[e];
              a[8 + e]  = (_Float16)x2[e];
              a[12 + e] = (_Float16)x3[e];
            }
          }
        }
        // ---- B fragments from LDS (2 x ds_load_b128 each) ----
        const _Float16* bp0 = bsh + ch * 512 + lane * 16;
        const _Float16* bp1 = bsh + (NCH + ch) * 512 + lane * 16;
        h8 p0 = *(const h8*)(bp0);
        h8 p1 = *(const h8*)(bp0 + 8);
        h8 q0 = *(const h8*)(bp1);
        h8 q1 = *(const h8*)(bp1 + 8);
        v16h bb0, bb1;
#pragma unroll
        for (int e = 0; e < 8; ++e) {
          bb0[e] = p0[e];
          bb0[8 + e] = p1[e];
          bb1[e] = q0[e];
          bb1[8 + e] = q1[e];
        }
        // (neg_a, A, neg_b, B, c_mod, C, reuse_a, reuse_b)
        c0 = __builtin_amdgcn_wmma_f32_16x16x32_f16(false, a, false, bb0,
                                                    (short)0, c0, false,
                                                    false);
        c1 = __builtin_amdgcn_wmma_f32_16x16x32_f16(false, a, false, bb1,
                                                    (short)0, c1, false,
                                                    false);
      }
      __syncthreads();  // protect bsh before next tap's staging
    }
  }

  const int co0 = tn0 * 16 + (lane & 15);
#pragma unroll
  for (int r = 0; r < 8; ++r) {
    int pm = tile_m * 16 + r + 8 * hi16;
    if (pm < npix) {
      float mval = mask[pm];
      long long obase = (long long)pm * out_c;
      float v0 = c0[r] * mval;
      float v1 = c1[r] * mval;
      if (accumulate) {
        out[obase + co0] += v0;
        out[obase + co0 + 16] += v1;
      } else {
        out[obase + co0] = v0;
        out[obase + co0 + 16] = v1;
      }
    }
  }
}

// ---------------------------------------------------------------------------
// Host orchestration
// ---------------------------------------------------------------------------
extern "C" void kernel_launch(void* const* d_in, const int* in_sizes, int n_in,
                              void* d_out, int out_size, void* d_ws,
                              size_t ws_size, hipStream_t stream) {
  (void)in_sizes; (void)n_in; (void)out_size; (void)ws_size;

  const float* depth = (const float*)d_in[0];
  float* out = (float*)d_out;
  float* ws = (float*)d_ws;

  const int B = 4, H0 = 352, W0 = 1216;
  const int Hs[3] = {176, 88, 44};
  const int Ws[3] = {608, 304, 152};
  const int OC[3] = {32, 64, 128};
  const int IC[3] = {1, 32, 64};
  const int KK[3] = {5, 5, 3};
  const int P[3]  = {B * 176 * 608, B * 88 * 304, B * 44 * 152};
  const long long OUTOFF[3] = {0,
                               (long long)P[0] * OC[0],
                               (long long)P[0] * OC[0] + (long long)P[1] * OC[1]};

  // params: d_in[1 + 14*lvl + j], dict insertion order per level:
  // 0 l1_w, 1 l2_w1, 2 l2_g1, 3 l2_b1, 4 l2_w2, 5 l3_w1, 6 l3_g1, 7 l3_b1,
  // 8 l3_w2, 9 l3_g2, 10 l3_b2, 11 l3_w3, 12 f_g, 13 f_b
  auto PRM = [&](int lvl, int j) -> const float* {
    return (const float*)d_in[1 + 14 * lvl + j];
  };

  // workspace layout (floats; every buffer stays 16B aligned)
  float* sumb   = ws;        // 256
  float* sumsqb = ws + 256;  // 256
  float* nbuf   = ws + 512;  // 1 (pad to 1024)
  float* masks[3];
  masks[0] = ws + 1024;
  masks[1] = masks[0] + P[0];
  masks[2] = masks[1] + P[1];
  float* xp = masks[2] + P[2];                   // pooled input
  const long long XPMAX = (long long)P[1] * 32;  // largest pooled x
  float* acc = xp + XPMAX;                       // branch sum
  const long long ACCMAX = (long long)P[0] * 32;
  float* t1 = acc + ACCMAX;  // branch temp 1
  const long long TMAX = (long long)P[0] * 64;
  float* t2 = t1 + TMAX;                    // branch temp 2
  _Float16* wswb = (_Float16*)(t2 + TMAX);  // swizzled weights (<=1.2MB)

  for (int lvl = 0; lvl < 3; ++lvl) {
    const int ic = IC[lvl], oc = OC[lvl], oc2 = 2 * oc, kk = KK[lvl];
    const int Hl = Hs[lvl], Wl = Ws[lvl];
    const int np = P[lvl];
    const float* xin = (lvl == 0) ? depth : (out + OUTOFF[lvl - 1]);
    const float* mprev = (lvl == 0) ? nullptr : masks[lvl - 1];
    const int Hi = (lvl == 0) ? H0 : Hs[lvl - 1];
    const int Wi = (lvl == 0) ? W0 : Ws[lvl - 1];

    // --- maxpool (+ mask) ---
    {
      long long total = (long long)np * ic;
      int blocks = (int)((total + 255) / 256);
      maxpool_kernel<<<blocks, 256, 0, stream>>>(xin, mprev, xp, masks[lvl],
                                                 B, Hi, Wi, Hl, Wl, ic);
    }
    mask_count_kernel<<<1, 256, 0, stream>>>(masks[lvl], np, nbuf);

    auto conv = [&](const float* xi, const float* wt, float* o, int icc,
                    int occ, int accum) {
      const int taps = kk * kk;
      const int nch = (icc + 31) / 32;
      long long wtot = (long long)taps * nch * (occ / 16) * 512;
      weight_swizzle_kernel<<<(int)((wtot + 255) / 256), 256, 0, stream>>>(
          wt, wswb, icc, occ, taps);
      int mt = (np + 15) / 16;
      dim3 grid((mt + 7) / 8, occ / 32);  // 2 n-tiles per wave
      if (kk == 5) {
        if (icc == 1)
          conv_wmma_kernel<1, 5><<<grid, 256, 0, stream>>>(
              xi, masks[lvl], wswb, o, np, Hl, Wl, occ, accum);
        else if (icc == 32)
          conv_wmma_kernel<32, 5><<<grid, 256, 0, stream>>>(
              xi, masks[lvl], wswb, o, np, Hl, Wl, occ, accum);
        else if (icc == 64)
          conv_wmma_kernel<64, 5><<<grid, 256, 0, stream>>>(
              xi, masks[lvl], wswb, o, np, Hl, Wl, occ, accum);
        else
          conv_wmma_kernel<128, 5><<<grid, 256, 0, stream>>>(
              xi, masks[lvl], wswb, o, np, Hl, Wl, occ, accum);
      } else {
        if (icc == 64)
          conv_wmma_kernel<64, 3><<<grid, 256, 0, stream>>>(
              xi, masks[lvl], wswb, o, np, Hl, Wl, occ, accum);
        else
          conv_wmma_kernel<256, 3><<<grid, 256, 0, stream>>>(
              xi, masks[lvl], wswb, o, np, Hl, Wl, occ, accum);
      }
    };
    auto bn = [&](float* xi, const float* g, const float* bt, float* o,
                  int C) {
      bn_reduce_kernel<<<C, 256, 0, stream>>>(xi, np, C, sumb, sumsqb);
      long long total = (long long)np * C;
      int blocks = (int)((total + 255) / 256);
      bn_norm_relu_kernel<<<blocks, 256, 0, stream>>>(
          xi, masks[lvl], g, bt, sumb, sumsqb, nbuf, o, np, C);
    };

    // --- layer1: conv (writes acc) ---
    conv(xp, PRM(lvl, 0), acc, ic, oc, 0);

    // --- layer2: conv -> BN+ReLU -> conv (accumulates into acc) ---
    conv(xp, PRM(lvl, 1), t1, ic, oc2, 0);
    bn(t1, PRM(lvl, 2), PRM(lvl, 3), t1, oc2);
    conv(t1, PRM(lvl, 4), acc, oc2, oc, 1);

    // --- layer3: (conv -> BN+ReLU) x2 -> conv (accumulates into acc) ---
    conv(xp, PRM(lvl, 5), t1, ic, oc2, 0);
    bn(t1, PRM(lvl, 6), PRM(lvl, 7), t1, oc2);
    conv(t1, PRM(lvl, 8), t2, oc2, oc2, 0);
    bn(t2, PRM(lvl, 9), PRM(lvl, 10), t2, oc2);
    conv(t2, PRM(lvl, 11), acc, oc2, oc, 1);

    // --- final BN + ReLU -> output slice (also next level's input) ---
    bn(acc, PRM(lvl, 12), PRM(lvl, 13), out + OUTOFF[lvl], oc);
  }
}
